// Model_85066122265068
// MI455X (gfx1250) — matrix operand order repaired and checked by reading, not compiled, not run on hardware
//
#include <hip/hip_runtime.h>
#include <hip/hip_bf16.h>
#include <math.h>

// Problem constants (from reference)
#define B_   8
#define G_   64
#define S_   256
#define DIM_ 41
#define ET_  128
#define H_   8
#define EK_  16
#define NH_  127

typedef __attribute__((ext_vector_type(16))) _Float16 v16h;
typedef __attribute__((ext_vector_type(8)))  _Float16 v8h;
typedef __attribute__((ext_vector_type(8)))  float    v8f;
typedef __attribute__((ext_vector_type(4)))  float    v4f;

// ---------------------------------------------------------------------------
// Pack A: fp32 (M x K) -> f16 (M x KP), zero-padded in k.
// ---------------------------------------------------------------------------
__global__ __launch_bounds__(256) void pack_a(const float* __restrict__ in,
                                              _Float16* __restrict__ out,
                                              int M, int K, int KP)
{
    const int idx = blockIdx.x * 256 + threadIdx.x;
    if (idx >= M * KP) return;
    const int r = idx / KP, k = idx % KP;
    out[idx] = (k < K) ? (_Float16)in[(size_t)r * K + k] : (_Float16)0.0f;
}

// ---------------------------------------------------------------------------
// Pack W: fp32 (K x N) -> f16 transposed (128 x KP), zero-padded in k and n.
// ---------------------------------------------------------------------------
__global__ __launch_bounds__(256) void pack_w(const float* __restrict__ in,
                                              _Float16* __restrict__ out,
                                              int K, int N, int KP)
{
    const int idx = blockIdx.x * 256 + threadIdx.x;
    if (idx >= 128 * KP) return;
    const int n = idx / KP, k = idx % KP;
    out[idx] = (k < K && n < N) ? (_Float16)in[(size_t)k * N + n] : (_Float16)0.0f;
}

// ---------------------------------------------------------------------------
// f16 WMMA GEMM on packed operands:
//   C(MxN,f32) = Ap(MxKP,f16) * WpT(128xKP,f16)^T + bias(N)
// KP is compile-time -> k-loop fully unrolls, all loads are b128, no guards.
// One wave per 16x16 output tile, 8 waves per block.
// ---------------------------------------------------------------------------
template <int KP>
__global__ __launch_bounds__(256) void gemm16(const _Float16* __restrict__ Ap,
                                              const _Float16* __restrict__ WpT,
                                              const float* __restrict__ bias,
                                              float* __restrict__ C,
                                              int M, int N)
{
    const int lane = threadIdx.x & 31;
    const int wave = threadIdx.x >> 5;
    const int m    = lane & 15;
    const int hi   = lane >> 4;
    const int tilesN = (N + 15) >> 4;
    const int tile = blockIdx.x * 8 + wave;
    if (tile >= (M >> 4) * tilesN) return;      // wave-uniform
    const int tm = (tile / tilesN) << 4;
    const int tn = (tile % tilesN) << 4;
    const int n  = tn + m;                      // < 128 always (WpT is padded)

    const _Float16* __restrict__ arow = Ap  + (size_t)(tm + m) * KP + (hi << 3);
    const _Float16* __restrict__ wrow = WpT + (size_t)n * KP        + (hi << 3);

    v8f acc = {};
#pragma unroll
    for (int kk = 0; kk < KP; kk += 32) {
        const v8h a0 = *(const v8h*)(arow + kk);
        const v8h a1 = *(const v8h*)(arow + kk + 16);
        const v8h b0 = *(const v8h*)(wrow + kk);
        const v8h b1 = *(const v8h*)(wrow + kk + 16);
        const v16h a = __builtin_shufflevector(a0, a1, 0,1,2,3,4,5,6,7,8,9,10,11,12,13,14,15);
        const v16h b = __builtin_shufflevector(b0, b1, 0,1,2,3,4,5,6,7,8,9,10,11,12,13,14,15);
        acc = __builtin_amdgcn_wmma_f32_16x16x32_f16(false, a, false, b,
                                                     (short)0, acc, false, false);
    }
    if (n < N) {
        const float bv = bias[n];
#pragma unroll
        for (int j = 0; j < 8; ++j) {
            C[(size_t)(tm + (hi << 3) + j) * N + n] = acc[j] + bv;
        }
    }
}

// ---------------------------------------------------------------------------
// scores[b,h,g,s] = (q[b,h,g,:16] . k[b,h,s,:16]) / 4  via WMMA (K=16 pad 32).
// Unconditional float4 loads; k>=16 half zeroed branchlessly.
// One block per (b,h); 8 waves cover the 4x16 = 64 tiles.
// ---------------------------------------------------------------------------
__global__ __launch_bounds__(256) void scores_wmma(const float* __restrict__ qproj,
                                                   const float* __restrict__ kproj,
                                                   float* __restrict__ scores)
{
    const int bh   = blockIdx.x;     // 0..63
    const int b    = bh / H_;
    const int h    = bh % H_;
    const int lane = threadIdx.x & 31;
    const int wave = threadIdx.x >> 5;
    const int m    = lane & 15;
    const int hi   = lane >> 4;

    for (int t = wave; t < 64; t += 8) {       // wave-uniform loop
        const int gt = (t >> 4) << 4;          // g-tile base
        const int st = (t & 15) << 4;          // s-tile base

        // A fragment: q row, k<16 only (i<8 slots); i>=8 slots are k>=16 -> 0
        const float* qrow = qproj + (size_t)(b * G_ + gt + m) * ET_ + h * EK_ + (hi << 3);
        const v4f q0 = *(const v4f*)(qrow);
        const v4f q1 = *(const v4f*)(qrow + 4);
        v16h a;
#pragma unroll
        for (int i = 0; i < 4; ++i) {
            a[i]      = (_Float16)q0[i];
            a[4 + i]  = (_Float16)q1[i];
            a[8 + i]  = (_Float16)0.0f;
            a[12 + i] = (_Float16)0.0f;
        }
        // B fragment: B[k][n=s] = kproj[s][h*16+k], k = hi*8 + i for i<8; i>=8 slots are k>=16 -> 0.
        // i>=8 slots need zeros; load unconditionally (ws slack keeps it
        // in-bounds) and zero by multiplying with 0.0f.
        const float* krow = kproj + (size_t)(b * S_ + st + m) * ET_ + h * EK_ + (hi << 3);
        const v4f k0 = *(const v4f*)(krow);
        const v4f k1 = *(const v4f*)(krow + 4);
        const v4f k2 = *(const v4f*)(krow + 8);
        const v4f k3 = *(const v4f*)(krow + 12);
        v16h bf;
#pragma unroll
        for (int i = 0; i < 4; ++i) {
            bf[i]      = (_Float16)k0[i];
            bf[4 + i]  = (_Float16)k1[i];
            bf[8 + i]  = (_Float16)(0.0f * k2[i]);
            bf[12 + i] = (_Float16)(0.0f * k3[i]);
        }
        v8f acc = {};
        acc = __builtin_amdgcn_wmma_f32_16x16x32_f16(false, a, false, bf,
                                                     (short)0, acc, false, false);
#pragma unroll
        for (int j = 0; j < 8; ++j) {
            scores[(size_t)(bh * G_ + gt + (hi << 3) + j) * S_ + st + m] = acc[j] * 0.25f;
        }
    }
}

// ---------------------------------------------------------------------------
// stride = sigmoid(stride_in @ Wr + br)
// ---------------------------------------------------------------------------
__global__ void stride_kernel(const float* __restrict__ sin_,
                              const float* __restrict__ Wr,
                              const float* __restrict__ br,
                              float* __restrict__ strideOut)
{
    const int d = threadIdx.x;
    if (d < DIM_) {
        float acc = br[d];
        for (int i = 0; i < DIM_; ++i) acc += sin_[i] * Wr[i * DIM_ + d];
        strideOut[d] = 1.0f / (1.0f + expf(-acc));
    }
}

// ---------------------------------------------------------------------------
// Masked per-d softmax over S + value contraction. Block = (b,h,g).
// ---------------------------------------------------------------------------
__device__ inline float waveMax(float v) {
#pragma unroll
    for (int off = 16; off > 0; off >>= 1) v = fmaxf(v, __shfl_xor(v, off, 32));
    return v;
}
__device__ inline float waveSum(float v) {
#pragma unroll
    for (int off = 16; off > 0; off >>= 1) v += __shfl_xor(v, off, 32);
    return v;
}

__global__ __launch_bounds__(256) void attn_kernel(const float* __restrict__ scores,
                                                   const float* __restrict__ value,
                                                   const float* __restrict__ mask,
                                                   const float* __restrict__ qt,
                                                   const float* __restrict__ tt,
                                                   const float* __restrict__ strideV,
                                                   float* __restrict__ xbuf)
{
    __shared__ float redA[8];
    __shared__ float redB[8];
    __shared__ float sstr[DIM_];

    const int idx = blockIdx.x;
    const int g = idx % G_;
    const int h = (idx / G_) % H_;
    const int b = idx / (G_ * H_);
    const int s = threadIdx.x;            // 0..255
    const int lane = s & 31, wid = s >> 5;

    if (s < DIM_) sstr[s] = strideV[s];
    __syncthreads();

    const float sc  = scores[(size_t)((b * H_ + h) * G_ + g) * S_ + s];
    const float ttv = tt[b * S_ + s];
    const float qg  = qt[g];

    for (int d = 0; d < DIM_; ++d) {
        const float sd = sstr[d];
        const float lo = qg - sd, hib = qg + sd;
        const bool msk = (mask[(size_t)(b * S_ + s) * DIM_ + d] != 0.0f)
                         && (ttv >= lo) && (ttv <= hib);
        const float val = msk ? sc : -1000000000.0f;

        // block max
        float wm = waveMax(val);
        if (lane == 0) redA[wid] = wm;
        __syncthreads();
        float mx = redA[0];
#pragma unroll
        for (int i = 1; i < 8; ++i) mx = fmaxf(mx, redA[i]);
        __syncthreads();

        const float e  = expf(val - mx);   // masked -> exactly 0.0f
        const float dv = e * value[(size_t)(b * S_ + s) * DIM_ + d];

        const float wse = waveSum(e);
        const float wsd = waveSum(dv);
        if (lane == 0) { redA[wid] = wse; redB[wid] = wsd; }
        __syncthreads();
        float se = 0.0f, sdot = 0.0f;
#pragma unroll
        for (int i = 0; i < 8; ++i) { se += redA[i]; sdot += redB[i]; }
        __syncthreads();

        if (threadIdx.x == 0) {
            xbuf[(size_t)(b * G_ + g) * (H_ * DIM_) + h * DIM_ + d] = sdot / se;
        }
    }
}

// ---------------------------------------------------------------------------
extern "C" void kernel_launch(void* const* d_in, const int* in_sizes, int n_in,
                              void* d_out, int out_size, void* d_ws, size_t ws_size,
                              hipStream_t stream)
{
    const float* query = (const float*)d_in[0];
    const float* key   = (const float*)d_in[1];
    const float* value = (const float*)d_in[2];
    const float* mask  = (const float*)d_in[3];
    const float* qt    = (const float*)d_in[4];
    const float* tt    = (const float*)d_in[5];
    const float* sin_  = (const float*)d_in[6];
    const float* Wq    = (const float*)d_in[7];
    const float* bq    = (const float*)d_in[8];
    const float* Wk    = (const float*)d_in[9];
    const float* bk    = (const float*)d_in[10];
    const float* Wo    = (const float*)d_in[11];
    const float* bo    = (const float*)d_in[12];
    const float* Wr    = (const float*)d_in[13];
    const float* br    = (const float*)d_in[14];
    float* out = (float*)d_out;

    // ---- workspace layout (fp32 region, then f16 region); slack between
    //      buffers covers the scores kernel's benign overreads.
    float* f32ws   = (float*)d_ws;
    float* strideV = f32ws;                                  // 64
    float* qproj   = strideV + 64;                           // 512*128   = 65536
    float* kproj   = qproj + 512 * 128;                      // 2048*128  = 262144
    float* scores  = kproj + 2048 * 128 + 64;                // 8*8*64*256= 1048576
    float* xbuf    = scores + (size_t)B_ * H_ * G_ * S_ + 64;// 512*328   = 167936

    _Float16* hws = (_Float16*)(xbuf + 512 * 328 + 64);
    _Float16* qAp = hws;                 // 512*128  f16
    _Float16* WqT = qAp + 512 * 128;     // 128*128
    _Float16* kAp = WqT + 128 * 128;     // 2048*64
    _Float16* WkT = kAp + 2048 * 64;     // 128*64
    _Float16* xAp = WkT + 128 * 64;      // 512*352
    _Float16* WoT = xAp + 512 * 352;     // 128*352
    // total ~7.1 MB

    stride_kernel<<<dim3(1), dim3(64), 0, stream>>>(sin_, Wr, br, strideV);

    // ---- q = query @ Wq + bq : (512x128)x(128x128), KP=128
    pack_a<<<dim3((512 * 128 + 255) / 256), dim3(256), 0, stream>>>(query, qAp, 512, 128, 128);
    pack_w<<<dim3((128 * 128 + 255) / 256), dim3(256), 0, stream>>>(Wq, WqT, 128, 128, 128);
    gemm16<128><<<dim3(32), dim3(256), 0, stream>>>(qAp, WqT, bq, qproj, 512, 128);

    // ---- k = key @ Wk + bk : (2048x41)x(41x128), KP=64
    pack_a<<<dim3((2048 * 64 + 255) / 256), dim3(256), 0, stream>>>(key, kAp, 2048, 41, 64);
    pack_w<<<dim3((128 * 64 + 255) / 256), dim3(256), 0, stream>>>(Wk, WkT, 41, 128, 64);
    gemm16<64><<<dim3(128), dim3(256), 0, stream>>>(kAp, WkT, bk, kproj, 2048, 128);

    // ---- scores per (b,h)
    scores_wmma<<<dim3(B_ * H_), dim3(256), 0, stream>>>(qproj, kproj, scores);

    // ---- masked softmax + value contraction
    attn_kernel<<<dim3(B_ * H_ * G_), dim3(256), 0, stream>>>(scores, value, mask,
                                                              qt, tt, strideV, xbuf);

    // ---- out = x @ Wo + bo : (512x328)x(328x127), KP=352
    pack_a<<<dim3((512 * 352 + 255) / 256), dim3(256), 0, stream>>>(xbuf, xAp, 512, 328, 352);
    pack_w<<<dim3((128 * 352 + 255) / 256), dim3(256), 0, stream>>>(Wo, WoT, 328, 127, 352);
    gemm16<352><<<dim3(32), dim3(256), 0, stream>>>(xAp, WoT, bo, out, 512, 127);
}
